// MultiHeadAttentionLayer_11639361372568
// MI455X (gfx1250) — compile-verified, matrix operand order checked by reading, not run on hardware
//
#include <hip/hip_runtime.h>
#include <hip/hip_bf16.h>

#define D_MODEL   1024
#define NUM_HEADS 16
#define D_K       64
#define B_SZ      2
#define SEQ       2048
#define NEG_BIG   (-1e12f)

typedef __bf16 bf16_t;
typedef __attribute__((ext_vector_type(16))) __bf16 v16bf;
typedef __attribute__((ext_vector_type(8)))  __bf16 v8bf;
typedef __attribute__((ext_vector_type(8)))  float  v8f;

// ---------------------------------------------------------------------------
// Fragment loaders per CDNA5 WMMA VGPR layout (ISA 7.12.2, 16-bit A 16x32):
//   lane half h (lane>>4), element e in [0,16):
//     k(e) = (e<8 ? 0 : 16) + h*8 + (e&7)
// A: lane&15 = M row.  B: lane&15 = N column, K-major per lane (same k map).
// ---------------------------------------------------------------------------
__device__ __forceinline__ v16bf load_frag_bf16(const bf16_t* __restrict__ p) {
    // p = &M[row][kbase + half*8]; two contiguous 16B chunks at +0 and +16 elems
    v8bf lo = *(const v8bf*)(p);
    v8bf hi = *(const v8bf*)(p + 16);
    v16bf f;
#pragma unroll
    for (int i = 0; i < 8; ++i) { f[i] = lo[i]; f[i + 8] = hi[i]; }
    return f;
}

__device__ __forceinline__ v16bf load_frag_f32(const float* __restrict__ p) {
    v16bf f;
#pragma unroll
    for (int i = 0; i < 8; ++i) {
        f[i]     = (bf16_t)p[i];
        f[i + 8] = (bf16_t)p[16 + i];
    }
    return f;
}

__device__ __forceinline__ v16bf load_frag_f32_nt(const float* __restrict__ p) {
    v16bf f;
#pragma unroll
    for (int i = 0; i < 8; ++i) {
        f[i]     = (bf16_t)__builtin_nontemporal_load(p + i);
        f[i + 8] = (bf16_t)__builtin_nontemporal_load(p + 16 + i);
    }
    return f;
}

__device__ __forceinline__ v8f wmma_bf16(v16bf a, v16bf b, v8f c) {
    return __builtin_amdgcn_wmma_f32_16x16x32_bf16(
        /*neg_a=*/false, a, /*neg_b=*/false, b,
        /*c_mod=*/(short)0, c, /*reuse_a=*/false, /*reuse_b=*/false);
}

// ---------------------------------------------------------------------------
// fp32 -> bf16 elementwise convert (weights)
// ---------------------------------------------------------------------------
__global__ void cvt_f32_bf16(const float* __restrict__ in, bf16_t* __restrict__ out, int n) {
    int i = blockIdx.x * blockDim.x + threadIdx.x;
    if (i < n) out[i] = (bf16_t)in[i];
}

// ---------------------------------------------------------------------------
// O = X[M,1024](f32) @ W[1024,1024](bf16)^T + bias ; M = B*S = 4096
// Block: 128 thr = 4 waves, each wave: 16(M) x 64(N), block tile 64x64.
// mode 0: out_bf [B,H,S,DK]     (Q, K)
// mode 1: out_bf [B,H,DK,S]     (V transposed for the A*V GEMM)
// mode 2: out_f  [M, D_MODEL]   (final x output)
// ---------------------------------------------------------------------------
__global__ void proj_gemm(const float*  __restrict__ X,
                          const bf16_t* __restrict__ W,
                          const float*  __restrict__ bias,
                          bf16_t* __restrict__ out_bf,
                          float*  __restrict__ out_f,
                          int mode) {
    const int lane = threadIdx.x & 31;
    const int wave = threadIdx.x >> 5;
    const int half = lane >> 4;
    const int l16  = lane & 15;

    const int tileN = blockIdx.x * 64;
    const int tileM = blockIdx.y * 64 + wave * 16;

    const float*  aptr = X + (size_t)(tileM + l16) * D_MODEL + half * 8;
    const bf16_t* bptr = W + (size_t)(tileN + l16) * D_MODEL + half * 8;

    v8f acc[4] = {};
    for (int k = 0; k < D_MODEL; k += 32) {
        v16bf a = load_frag_f32(aptr + k);
#pragma unroll
        for (int j = 0; j < 4; ++j) {
            v16bf b = load_frag_bf16(bptr + (size_t)j * 16 * D_MODEL + k);
            acc[j] = wmma_bf16(a, b, acc[j]);
        }
    }

#pragma unroll
    for (int j = 0; j < 4; ++j) {
        const int n  = tileN + j * 16 + l16;
        const float bv = bias[n];
#pragma unroll
        for (int r = 0; r < 8; ++r) {
            const int m = tileM + half * 8 + r;
            const float v = acc[j][r] + bv;
            if (mode == 2) {
                out_f[(size_t)m * D_MODEL + n] = v;
            } else {
                const int b_ = m / SEQ, s_ = m % SEQ;
                const int h_ = n / D_K, d_ = n % D_K;
                size_t idx;
                if (mode == 0)
                    idx = ((size_t)(b_ * NUM_HEADS + h_) * SEQ + s_) * D_K + d_;
                else
                    idx = ((size_t)(b_ * NUM_HEADS + h_) * D_K + d_) * SEQ + s_;
                out_bf[idx] = (bf16_t)v;
            }
        }
    }
}

// ---------------------------------------------------------------------------
// scores[bh,q,k] = (Q[bh,q,:] . K[bh,k,:]) / 8, masked; NT-store into d_out.
// ---------------------------------------------------------------------------
__global__ void scores_kernel(const bf16_t* __restrict__ Q,   // [B*H, S, DK]
                              const bf16_t* __restrict__ Km,  // [B*H, S, DK]
                              const unsigned char* __restrict__ mask, // [B, S]
                              float* __restrict__ scores) {   // [B*H, S, S]
    const int bh = blockIdx.z;
    const int b_ = bh / NUM_HEADS;
    const int lane = threadIdx.x & 31;
    const int wave = threadIdx.x >> 5;
    const int half = lane >> 4;
    const int l16  = lane & 15;

    const int tileQ = blockIdx.y * 64 + wave * 16;
    const int tileK = blockIdx.x * 64;

    const bf16_t* aptr = Q  + (size_t)bh * SEQ * D_K + (size_t)(tileQ + l16) * D_K + half * 8;
    const bf16_t* bptr = Km + (size_t)bh * SEQ * D_K + (size_t)(tileK + l16) * D_K + half * 8;

    v8f acc[4] = {};
#pragma unroll
    for (int k = 0; k < D_K; k += 32) {
        v16bf a = load_frag_bf16(aptr + k);
#pragma unroll
        for (int j = 0; j < 4; ++j) {
            v16bf b = load_frag_bf16(bptr + (size_t)j * 16 * D_K + k);
            acc[j] = wmma_bf16(a, b, acc[j]);
        }
    }

    const float scale = 0.125f; // 1/sqrt(D_K)
    float* srow = scores + (size_t)bh * SEQ * SEQ;
#pragma unroll
    for (int j = 0; j < 4; ++j) {
        const int kc = tileK + j * 16 + l16;
        const bool ok = mask[b_ * SEQ + kc] != 0;
#pragma unroll
        for (int r = 0; r < 8; ++r) {
            const int q = tileQ + half * 8 + r;
            const float v = ok ? acc[j][r] * scale : NEG_BIG;
            __builtin_nontemporal_store(v, &srow[(size_t)q * SEQ + kc]);
        }
    }
}

// ---------------------------------------------------------------------------
// In-place row softmax over 2048-wide rows. 256 threads, 8 elems/thread.
// ---------------------------------------------------------------------------
__global__ void softmax_kernel(float* __restrict__ attn) {
    float* p = attn + (size_t)blockIdx.x * SEQ;
    const int tid = threadIdx.x;

    float v[8];
    float lmax = -3.4e38f;
#pragma unroll
    for (int i = 0; i < 8; ++i) {
        v[i] = p[tid + i * 256];
        lmax = fmaxf(lmax, v[i]);
    }
#pragma unroll
    for (int off = 16; off > 0; off >>= 1)
        lmax = fmaxf(lmax, __shfl_xor(lmax, off, 32));

    __shared__ float smax[8], ssum[8];
    if ((tid & 31) == 0) smax[tid >> 5] = lmax;
    __syncthreads();
    float rmax = smax[0];
#pragma unroll
    for (int i = 1; i < 8; ++i) rmax = fmaxf(rmax, smax[i]);

    float lsum = 0.f;
#pragma unroll
    for (int i = 0; i < 8; ++i) { v[i] = __expf(v[i] - rmax); lsum += v[i]; }
#pragma unroll
    for (int off = 16; off > 0; off >>= 1)
        lsum += __shfl_xor(lsum, off, 32);
    if ((tid & 31) == 0) ssum[tid >> 5] = lsum;
    __syncthreads();
    float rsum = 0.f;
#pragma unroll
    for (int i = 0; i < 8; ++i) rsum += ssum[i];

    const float inv = 1.0f / rsum;
#pragma unroll
    for (int i = 0; i < 8; ++i) p[tid + i * 256] = v[i] * inv;
}

// ---------------------------------------------------------------------------
// ctx[b,q,h*64+d] = sum_k attn[bh,q,k] * V[bh,k,d]   (V stored transposed)
// M = S (q), N = 64 (d), K = S. One N-block (4 j-frags cover D_K).
// ---------------------------------------------------------------------------
__global__ void context_kernel(const float*  __restrict__ attn, // [B*H, S, S]
                               const bf16_t* __restrict__ Vt,   // [B*H, DK, S]
                               float* __restrict__ ctx) {       // [B, S, D_MODEL]
    const int bh = blockIdx.z;
    const int b_ = bh / NUM_HEADS, h_ = bh % NUM_HEADS;
    const int lane = threadIdx.x & 31;
    const int wave = threadIdx.x >> 5;
    const int half = lane >> 4;
    const int l16  = lane & 15;

    const int tileQ = blockIdx.y * 64 + wave * 16;
    const float*  aptr = attn + (size_t)bh * SEQ * SEQ + (size_t)(tileQ + l16) * SEQ + half * 8;
    const bf16_t* bptr = Vt   + (size_t)bh * D_K * SEQ + (size_t)l16 * SEQ + half * 8;

    v8f acc[4] = {};
    for (int k = 0; k < SEQ; k += 32) {
        v16bf a = load_frag_f32_nt(aptr + k);
#pragma unroll
        for (int j = 0; j < 4; ++j) {
            v16bf b = load_frag_bf16(bptr + (size_t)j * 16 * SEQ + k);
            acc[j] = wmma_bf16(a, b, acc[j]);
        }
    }

#pragma unroll
    for (int j = 0; j < 4; ++j) {
        const int d = j * 16 + l16;
#pragma unroll
        for (int r = 0; r < 8; ++r) {
            const int q = tileQ + half * 8 + r;
            ctx[((size_t)(b_ * SEQ + q)) * D_MODEL + h_ * D_K + d] = acc[j][r];
        }
    }
}

// ---------------------------------------------------------------------------
extern "C" void kernel_launch(void* const* d_in, const int* in_sizes, int n_in,
                              void* d_out, int out_size, void* d_ws, size_t ws_size,
                              hipStream_t stream) {
    const float* query = (const float*)d_in[0];
    const float* key_  = (const float*)d_in[1];
    const float* value = (const float*)d_in[2];
    const unsigned char* mask = (const unsigned char*)d_in[3];
    const float* WQ_w = (const float*)d_in[4];
    const float* WQ_b = (const float*)d_in[5];
    const float* WK_w = (const float*)d_in[6];
    const float* WK_b = (const float*)d_in[7];
    const float* WV_w = (const float*)d_in[8];
    const float* WV_b = (const float*)d_in[9];
    const float* WO_w = (const float*)d_in[10];
    const float* WO_b = (const float*)d_in[11];

    float* out_x    = (float*)d_out;
    float* out_attn = out_x + (size_t)B_SZ * SEQ * D_MODEL;

    char* ws = (char*)d_ws;
    size_t o = 0;
    const size_t W_ELEMS = (size_t)D_MODEL * D_MODEL;           // 1M
    const size_t P_ELEMS = (size_t)B_SZ * SEQ * D_MODEL;        // 4M
    bf16_t* wq  = (bf16_t*)(ws + o); o += W_ELEMS * 2;
    bf16_t* wk  = (bf16_t*)(ws + o); o += W_ELEMS * 2;
    bf16_t* wv  = (bf16_t*)(ws + o); o += W_ELEMS * 2;
    bf16_t* wo  = (bf16_t*)(ws + o); o += W_ELEMS * 2;
    bf16_t* Qp  = (bf16_t*)(ws + o); o += P_ELEMS * 2;          // [B,H,S,DK]
    bf16_t* Kp  = (bf16_t*)(ws + o); o += P_ELEMS * 2;          // [B,H,S,DK]
    bf16_t* Vt  = (bf16_t*)(ws + o); o += P_ELEMS * 2;          // [B,H,DK,S]
    float*  ctx = (float*) (ws + o); o += P_ELEMS * 4;          // [B,S,D_MODEL]

    // 1) weights -> bf16
    const int nw = (int)W_ELEMS;
    cvt_f32_bf16<<<(nw + 255) / 256, 256, 0, stream>>>(WQ_w, wq, nw);
    cvt_f32_bf16<<<(nw + 255) / 256, 256, 0, stream>>>(WK_w, wk, nw);
    cvt_f32_bf16<<<(nw + 255) / 256, 256, 0, stream>>>(WV_w, wv, nw);
    cvt_f32_bf16<<<(nw + 255) / 256, 256, 0, stream>>>(WO_w, wo, nw);

    // 2) Q/K/V projections (WMMA bf16)
    dim3 blk(128);
    dim3 gProj(D_MODEL / 64, (B_SZ * SEQ) / 64);
    proj_gemm<<<gProj, blk, 0, stream>>>(query, wq, WQ_b, Qp, nullptr, 0);
    proj_gemm<<<gProj, blk, 0, stream>>>(key_,  wk, WK_b, Kp, nullptr, 0);
    proj_gemm<<<gProj, blk, 0, stream>>>(value, wv, WV_b, Vt, nullptr, 1);

    // 3) scores = QK^T/8 (masked), NT-stored into d_out attention region
    dim3 gSc(SEQ / 64, SEQ / 64, B_SZ * NUM_HEADS);
    scores_kernel<<<gSc, blk, 0, stream>>>(Qp, Kp, mask, out_attn);

    // 4) in-place softmax over the 537 MB attention matrix
    softmax_kernel<<<B_SZ * NUM_HEADS * SEQ, 256, 0, stream>>>(out_attn);

    // 5) ctx = attn @ V
    dim3 gCtx(1, SEQ / 64, B_SZ * NUM_HEADS);
    context_kernel<<<gCtx, blk, 0, stream>>>(out_attn, Vt, ctx);

    // 6) x = ctx @ WO^T + b -> d_out
    proj_gemm<<<gProj, blk, 0, stream>>>(ctx, wo, WO_b, nullptr, out_x, 2);
}
